// VisualEncoder_62766652063762
// MI455X (gfx1250) — matrix operand order repaired and checked by reading, not compiled, not run on hardware
//
#include <hip/hip_runtime.h>

// ---------------------------------------------------------------------------
// MI455X (gfx1250) ViT encoder, bf16 WMMA (v_wmma_f32_16x16x32_bf16) everywhere.
// Compute-bound (~0.72 TFLOP vs ~150MB traffic) -> matrix pipes + bf16 weights
// resident in the 192MB L2. Attention computes S^T = K*Q^T so the softmax'd
// P registers are directly in A-fragment order (no LDS re-layout).
// ---------------------------------------------------------------------------

typedef __attribute__((ext_vector_type(16))) __bf16 v16bf;
typedef __attribute__((ext_vector_type(8)))  float   v8f;
typedef __attribute__((ext_vector_type(8)))  __bf16  bf16x8;

#define DEV static __device__ __forceinline__

static constexpr int kB  = 64;
static constexpr int kL  = 257;          // 256 patches + cls
static constexpr int kD  = 512;
static constexpr int kNH = 8;
static constexpr int kNL = 6;
static constexpr int kDM = 2048;
static constexpr int kDO = 512;
static constexpr int kM  = kB * kL;      // 16448 tokens
static constexpr int kMP = kB * 256;     // 16384 patch rows
static constexpr int kKP = 768;          // C*P*P
static constexpr int kVL = 296;          // LDS V^T row stride (keys, padded)

DEV int imin(int a, int b) { return a < b ? a : b; }

DEV v8f wmma_bf16(v16bf a, v16bf b, v8f c) {
  return __builtin_amdgcn_wmma_f32_16x16x32_bf16(false, a, false, b, (short)0, c,
                                                 false, false);
}

DEV float bcast_lane(float v, int srcLane) {  // uniform-lane broadcast: v_readlane
  return __int_as_float(__builtin_amdgcn_readlane(__float_as_int(v), srcLane));
}

// A fragment (16M x 32K) from row-major bf16. lane m = lane&15, half = lane>>4;
// elements 0..7 hold K = half*8+0..7, elements 8..15 hold K = 16+half*8+0..7.
DEV v16bf load_a_frag(const __bf16* base, int lda, int row, int kk, int half) {
  const __bf16* p = base + (size_t)row * lda + kk + half * 8;
  bf16x8 lo = *(const bf16x8*)p;
  bf16x8 hi = *(const bf16x8*)(p + 16);
  v16bf a;
#pragma unroll
  for (int i = 0; i < 8; ++i) { a[i] = lo[i]; a[i + 8] = hi[i]; }
  return a;
}

// B fragment (32K x 16N) from N-major bf16 (ldb = K elements per N row).
// Lane n = lane&15, elements 0..7 hold K = half*8+0..7, 8..15 hold K = 16+half*8+0..7.
DEV v16bf load_b_frag(const __bf16* base, int ldb, int nrow, int kk, int half) {
  const __bf16* p = base + (size_t)nrow * ldb + kk + half * 8;
  bf16x8 lo = *(const bf16x8*)p;
  bf16x8 hi = *(const bf16x8*)(p + 16);
  v16bf b;
#pragma unroll
  for (int i = 0; i < 8; ++i) { b[i] = lo[i]; b[i + 8] = hi[i]; }
  return b;
}

DEV float gelu_exact(float x) { return 0.5f * x * (1.0f + erff(x * 0.70710678118654752f)); }

// ---------------------------------------------------------------------------
// Generic WMMA GEMM: C[M,N] = A[M,K](bf16) * Wt[N,K](bf16)^T + bias, epilogues:
// MODE 0: f32 store | 1: bf16 store | 2: bf16 gelu | 3: f32 residual add
// MODE 4: patch-embed (adds pos emb, scatters rows past the cls token)
// ---------------------------------------------------------------------------
template <int MODE, bool FULL>
DEV void gemm_epilogue(const v8f (&c)[2][4], const float* __restrict__ bias,
                       float* outF, __bf16* outB, const float* resid,
                       const float* __restrict__ pos, int m0, int n0, int col,
                       int half, int M, int N) {
#pragma unroll
  for (int mf = 0; mf < 2; ++mf) {
#pragma unroll
    for (int nf = 0; nf < 4; ++nf) {
      const int n = n0 + nf * 16 + col;
#pragma unroll
      for (int r = 0; r < 8; ++r) {
        const int m = m0 + mf * 16 + r + 8 * half;
        if (FULL || m < M) {
          float v = c[mf][nf][r] + bias[n];
          if (MODE == 0) outF[(size_t)m * N + n] = v;
          if (MODE == 1) outB[(size_t)m * N + n] = (__bf16)v;
          if (MODE == 2) outB[(size_t)m * N + n] = (__bf16)gelu_exact(v);
          if (MODE == 3) outF[(size_t)m * N + n] = resid[(size_t)m * N + n] + v;
          if (MODE == 4) {
            int bi = m >> 8, pi = m & 255;
            outF[((size_t)bi * kL + 1 + pi) * kD + n] = v + pos[(size_t)pi * kD + n];
          }
        }
      }
    }
  }
}

template <int MODE>
__global__ __launch_bounds__(256) void gemm_wmma_kernel(
    const __bf16* __restrict__ A, const __bf16* __restrict__ Wt,
    const float* __restrict__ bias, float* outF, __bf16* outB,
    const float* resid, const float* __restrict__ pos, int M, int N, int K) {
  const int lane = threadIdx.x & 31;
  const int wv   = threadIdx.x >> 5;
  const int half = lane >> 4;
  const int col  = lane & 15;
  const int m0 = blockIdx.x * 128 + (wv & 3) * 32;   // 4 waves along M
  const int n0 = blockIdx.y * 128 + (wv >> 2) * 64;  // 2 waves along N
  const __bf16* pA0 = A + (size_t)imin(m0 + col, M - 1) * K;
  const __bf16* pA1 = A + (size_t)imin(m0 + 16 + col, M - 1) * K;
  const __bf16* pB  = Wt + (size_t)(n0 + col) * K;
  v8f c[2][4];
#pragma unroll
  for (int i = 0; i < 2; ++i)
#pragma unroll
    for (int j = 0; j < 4; ++j)
#pragma unroll
      for (int r = 0; r < 8; ++r) c[i][j][r] = 0.0f;

  for (int kk = 0; kk < K; kk += 32) {
    __builtin_prefetch(pA0 + kk + 64, 0, 1);
    v16bf a0 = load_a_frag(pA0, K, 0, kk, half);
    v16bf a1 = load_a_frag(pA1, K, 0, kk, half);
#pragma unroll
    for (int nf = 0; nf < 4; ++nf) {
      v16bf b = load_b_frag(pB, K, nf * 16, kk, half);
      c[0][nf] = wmma_bf16(a0, b, c[0][nf]);
      c[1][nf] = wmma_bf16(a1, b, c[1][nf]);
    }
  }
  if (m0 + 32 <= M)  // uniform branch: full tile needs no store guards
    gemm_epilogue<MODE, true>(c, bias, outF, outB, resid, pos, m0, n0, col, half, M, N);
  else
    gemm_epilogue<MODE, false>(c, bias, outF, outB, resid, pos, m0, n0, col, half, M, N);
}

// ---------------------------------------------------------------------------
// Flash attention, one workgroup per (batch, head), wave per 16-query tile.
// Scores are computed transposed (S^T = K*Q^T): lane = query column, register
// = key. The softmax'd probabilities are then *already* in A-fragment element
// order for the P*V WMMA (elem 0..7 -> key half*8+i from sa, elem 8..15 ->
// key 16+half*8+i from sb), so no cross-lane re-layout is needed. V^T is
// staged once per workgroup in LDS for contiguous B-fragment reads.
// TAIL: clamp/mask padded keys (only the last of 9 steps). FIRST: skip the
// (zero-valued) accumulator rescale of the online softmax.
// ---------------------------------------------------------------------------
template <bool TAIL, bool FIRST>
DEV void attn_step(int kb0, const __bf16* kbase, const __bf16* vTp,
                   v16bf bq0, v16bf bq1, int half, int col,
                   float& mq, float& lq, v8f (&o)[4]) {
  const float scale = 0.125f;  // 1/sqrt(64)
  int rkA = kb0 + col;
  int rkB = kb0 + 16 + col;
  if (TAIL) { rkA = imin(rkA, kL - 1); rkB = imin(rkB, kL - 1); }
  v8f sa, sb;
#pragma unroll
  for (int r = 0; r < 8; ++r) { sa[r] = 0.0f; sb[r] = 0.0f; }
  {
    v16bf ak;
    ak = load_a_frag(kbase, 1536, rkA, 0,  half); sa = wmma_bf16(ak, bq0, sa);
    ak = load_a_frag(kbase, 1536, rkA, 32, half); sa = wmma_bf16(ak, bq1, sa);
    ak = load_a_frag(kbase, 1536, rkB, 0,  half); sb = wmma_bf16(ak, bq0, sb);
    ak = load_a_frag(kbase, 1536, rkB, 32, half); sb = wmma_bf16(ak, bq1, sb);
  }
  // key of sa[r] = kb0+half*8+r, of sb[r] = kb0+16+half*8+r
  float S[16], tm = -1e30f;
#pragma unroll
  for (int r = 0; r < 8; ++r) {
    float va = sa[r] * scale;
    float vb = sb[r] * scale;
    if (TAIL) {
      const int ka = kb0 + half * 8 + r;
      if (ka >= kL)      va = -1e30f;
      if (ka + 16 >= kL) vb = -1e30f;
    }
    S[r] = va; S[r + 8] = vb;
    tm = fmaxf(tm, fmaxf(va, vb));
  }
  tm = fmaxf(tm, __shfl_xor(tm, 16, 32));  // combine key halves
  const float mn = FIRST ? tm : fmaxf(mq, tm);
  float ps = 0.0f;
  v16bf ap;  // P directly in A-fragment element order
#pragma unroll
  for (int i = 0; i < 16; ++i) {
    float p = __expf(S[i] - mn);
    ps += p;
    ap[i] = (__bf16)p;
  }
  ps += __shfl_xor(ps, 16, 32);
  if (FIRST) {
    mq = mn;
    lq = ps;
  } else {
    const float sc = __expf(mq - mn);
    mq = mn;
    lq = lq * sc + ps;
    // rescale o: register r holds query row r+8*half; fetch its sc via readlane
#pragma unroll
    for (int r = 0; r < 8; ++r) {
      float s0 = bcast_lane(sc, r);
      float s1 = bcast_lane(sc, r + 8);
      float scr = half ? s1 : s0;
#pragma unroll
      for (int nf = 0; nf < 4; ++nf) o[nf][r] *= scr;
    }
  }
  // P @ V: B fragments from LDS V^T (contiguous key runs per d row)
#pragma unroll
  for (int nf = 0; nf < 4; ++nf) {
    v16bf bv = load_b_frag(vTp, kVL, nf * 16 + col, kb0, half);
    o[nf] = wmma_bf16(ap, bv, o[nf]);
  }
}

__global__ __launch_bounds__(256) void attn_kernel(const __bf16* __restrict__ qkv,
                                                   __bf16* __restrict__ out) {
  __shared__ __bf16 vT[64][kVL];   // 64 d x 296 keys (zero-padded) = 37.9KB
  const int lane = threadIdx.x & 31;
  const int wv   = threadIdx.x >> 5;
  const int half = lane >> 4;
  const int col  = lane & 15;
  const int b = blockIdx.x >> 3;
  const int h = blockIdx.x & 7;
  const __bf16* qbase = qkv + (size_t)b * kL * 1536 + h * 64;
  const __bf16* kbase = qbase + 512;
  const __bf16* vbase = qbase + 1024;

  // zero the padded key tail (keys 257..295) so padded P*V terms are exact 0
  for (int i = threadIdx.x; i < 64 * 64; i += 256) {
    int d = i >> 6, key = 257 + (i & 63);
    if (key < kVL) vT[d][key] = (__bf16)0.0f;
  }
  // cooperative V^T staging: vectorized global reads, transposed LDS writes
  for (int i = threadIdx.x; i < kL * 8; i += 256) {
    int key = i >> 3, dc = (i & 7) * 8;
    bf16x8 v8 = *(const bf16x8*)(vbase + (size_t)key * 1536 + dc);
#pragma unroll
    for (int e = 0; e < 8; ++e) vT[dc + e][key] = v8[e];
  }
  __syncthreads();
  const __bf16* vTp = &vT[0][0];

  for (int t = wv; t < 17; t += 8) {  // 17 query tiles of 16 (L=257 padded)
    const int qrow = imin(t * 16 + col, kL - 1);
    v16bf bq0 = load_b_frag(qbase, 1536, qrow, 0, half);   // Q as B operand
    v16bf bq1 = load_b_frag(qbase, 1536, qrow, 32, half);
    float mq = -1e30f, lq = 0.0f;  // per-lane state for query `col`
    v8f o[4];
#pragma unroll
    for (int nf = 0; nf < 4; ++nf)
#pragma unroll
      for (int r = 0; r < 8; ++r) o[nf][r] = 0.0f;

    attn_step<false, true>(0, kbase, vTp, bq0, bq1, half, col, mq, lq, o);
    for (int j = 1; j < 8; ++j)  // keys 32..255: no clamping, no masking
      attn_step<false, false>(j * 32, kbase, vTp, bq0, bq1, half, col, mq, lq, o);
    attn_step<true, false>(256, kbase, vTp, bq0, bq1, half, col, mq, lq, o);

    // normalize and store: o register r = query t*16+r+8*half, lane col = d lane
#pragma unroll
    for (int r = 0; r < 8; ++r) {
      float l0 = bcast_lane(lq, r);
      float l1 = bcast_lane(lq, r + 8);
      float rinv = 1.0f / (half ? l1 : l0);
      const int q = t * 16 + r + 8 * half;
      if (q < kL) {
#pragma unroll
        for (int nf = 0; nf < 4; ++nf)
          out[((size_t)b * kL + q) * kD + h * 64 + nf * 16 + col] =
              (__bf16)(o[nf][r] * rinv);
      }
    }
  }
}

// ---------------------------------------------------------------------------
// LayerNorm over D=512, wave per row, shuffle reductions.
// ---------------------------------------------------------------------------
template <bool OUTF32>
__global__ __launch_bounds__(256) void layernorm_kernel(
    const float* __restrict__ X, const float* __restrict__ g,
    const float* __restrict__ be, float* outF, __bf16* outB, int M) {
  const int row = blockIdx.x * 8 + (threadIdx.x >> 5);
  const int lane = threadIdx.x & 31;
  if (row >= M) return;
  const float* xr = X + (size_t)row * kD;
  float v[16], s = 0.f, s2 = 0.f;
#pragma unroll
  for (int i = 0; i < 16; ++i) {
    v[i] = xr[lane + i * 32];
    s += v[i]; s2 += v[i] * v[i];
  }
#pragma unroll
  for (int mm = 1; mm < 32; mm <<= 1) {
    s  += __shfl_xor(s, mm, 32);
    s2 += __shfl_xor(s2, mm, 32);
  }
  const float mean = s * (1.0f / kD);
  const float var  = s2 * (1.0f / kD) - mean * mean;
  const float rs   = rsqrtf(var + 1e-5f);
#pragma unroll
  for (int i = 0; i < 16; ++i) {
    const int idx = lane + i * 32;
    const float ov = (v[i] - mean) * rs * g[idx] + be[idx];
    if (OUTF32) outF[(size_t)row * kD + idx] = ov;
    else        outB[(size_t)row * kD + idx] = (__bf16)ov;
  }
}

// ------------------------- small utility kernels ---------------------------
__global__ void transpose_bf16_kernel(const float* __restrict__ W,
                                      __bf16* __restrict__ Wt, int K, int N) {
  size_t idx = (size_t)blockIdx.x * 256 + threadIdx.x;
  if (idx >= (size_t)K * N) return;
  int k = (int)(idx / N), n = (int)(idx % N);
  Wt[(size_t)n * K + k] = (__bf16)W[idx];
}

__global__ void pos_kernel(const float* __restrict__ row_emb,
                           const float* __restrict__ col_emb,
                           float* __restrict__ pos) {
  int idx = blockIdx.x * 256 + threadIdx.x;  // 256*512
  if (idx >= 256 * kD) return;
  int p = idx >> 9, n = idx & 511;
  int ph = p >> 6, pw = p & 63;
  pos[idx] = (n < 256) ? row_emb[ph * 256 + n] : col_emb[pw * 256 + (n - 256)];
}

__global__ void cls_kernel(const float* __restrict__ cls, float* __restrict__ x) {
  int idx = blockIdx.x * 256 + threadIdx.x;  // 64*512
  if (idx >= kB * kD) return;
  int b = idx >> 9, n = idx & 511;
  x[(size_t)b * kL * kD + n] = cls[n];
}

__global__ void patchify_kernel(const float* __restrict__ px,
                                __bf16* __restrict__ A0) {
  size_t idx = (size_t)blockIdx.x * 256 + threadIdx.x;  // 16384*768
  if (idx >= (size_t)kMP * kKP) return;
  int q  = (int)(idx % kKP);
  int mp = (int)(idx / kKP);
  int b = mp >> 8, p = mp & 255;
  int c = q >> 8, t = q & 255;
  int py = t >> 4, pxi = t & 15;
  int ph = p >> 6, pw = p & 63;
  size_t src = (((size_t)b * 3 + c) * 64 + ph * 16 + py) * 1024 + pw * 16 + pxi;
  A0[idx] = (__bf16)px[src];
}

__global__ void vismask_kernel(int* __restrict__ m, int count) {
  int idx = blockIdx.x * 256 + threadIdx.x;
  if (idx < count) m[idx] = 1;
}

// ---------------------------------------------------------------------------
extern "C" void kernel_launch(void* const* d_in, const int* in_sizes, int n_in,
                              void* d_out, int out_size, void* d_ws, size_t ws_size,
                              hipStream_t stream) {
  const float* pixel   = (const float*)d_in[0];
  const float* patch_w = (const float*)d_in[1];
  const float* patch_b = (const float*)d_in[2];
  const float* row_emb = (const float*)d_in[3];
  const float* col_emb = (const float*)d_in[4];
  const float* cls     = (const float*)d_in[5];
  const float* ln1_s   = (const float*)d_in[6];
  const float* ln1_b   = (const float*)d_in[7];
  const float* qkv_w   = (const float*)d_in[8];
  const float* qkv_b   = (const float*)d_in[9];
  const float* out_w   = (const float*)d_in[10];
  const float* out_b   = (const float*)d_in[11];
  const float* ln2_s   = (const float*)d_in[12];
  const float* ln2_b   = (const float*)d_in[13];
  const float* mlp_w1  = (const float*)d_in[14];
  const float* mlp_b1  = (const float*)d_in[15];
  const float* mlp_w2  = (const float*)d_in[16];
  const float* mlp_b2  = (const float*)d_in[17];
  const float* lnf_s   = (const float*)d_in[18];
  const float* lnf_b   = (const float*)d_in[19];
  const float* br_w1   = (const float*)d_in[20];
  const float* br_b1   = (const float*)d_in[21];
  const float* br_w2   = (const float*)d_in[22];
  const float* br_b2   = (const float*)d_in[23];
  const float* brln_s  = (const float*)d_in[24];
  const float* brln_b  = (const float*)d_in[25];
  (void)in_sizes; (void)n_in; (void)out_size; (void)ws_size;

  char* wp = (char*)d_ws;
  auto alloc = [&](size_t bytes) {
    char* p = wp; wp += (bytes + 255) & ~(size_t)255; return p;
  };
  __bf16* WtP  = (__bf16*)alloc((size_t)kD * kKP * 2);
  __bf16* Wtq  = (__bf16*)alloc((size_t)kNL * 3 * kD * kD * 2);
  __bf16* Wto  = (__bf16*)alloc((size_t)kNL * kD * kD * 2);
  __bf16* Wtm1 = (__bf16*)alloc((size_t)kNL * kDM * kD * 2);
  __bf16* Wtm2 = (__bf16*)alloc((size_t)kNL * kD * kDM * 2);
  __bf16* WtB1 = (__bf16*)alloc((size_t)1024 * kD * 2);
  __bf16* WtB2 = (__bf16*)alloc((size_t)kDO * 1024 * 2);
  float*  posb = (float*)alloc((size_t)256 * kD * 4);
  __bf16* A0   = (__bf16*)alloc((size_t)kMP * kKP * 2);
  float*  x    = (float*)alloc((size_t)kM * kD * 4);
  __bf16* xn   = (__bf16*)alloc((size_t)kM * kD * 2);
  __bf16* qkvb = (__bf16*)alloc((size_t)kM * 3 * kD * 2);
  __bf16* atno = (__bf16*)alloc((size_t)kM * kD * 2);
  __bf16* hb   = (__bf16*)alloc((size_t)kM * kDM * 2);
  float*  h2 = (float*)qkvb;  // reuse: qkv dead after layers (needs kM*2048B)
  __bf16* h1 = hb;            // reuse: mlp hidden buffer for bridge hidden

  auto cdiv = [](size_t a, size_t b) { return (unsigned)((a + b - 1) / b); };
  dim3 blk(256);

  // --- weight prep: fp32 -> bf16, transposed to N-major for B-fragments ---
  transpose_bf16_kernel<<<cdiv((size_t)kKP * kD, 256), blk, 0, stream>>>(patch_w, WtP, kKP, kD);
  for (int l = 0; l < kNL; ++l) {
    transpose_bf16_kernel<<<cdiv((size_t)kD * 3 * kD, 256), blk, 0, stream>>>(
        qkv_w + (size_t)l * kD * 3 * kD, Wtq + (size_t)l * 3 * kD * kD, kD, 3 * kD);
    transpose_bf16_kernel<<<cdiv((size_t)kD * kD, 256), blk, 0, stream>>>(
        out_w + (size_t)l * kD * kD, Wto + (size_t)l * kD * kD, kD, kD);
    transpose_bf16_kernel<<<cdiv((size_t)kD * kDM, 256), blk, 0, stream>>>(
        mlp_w1 + (size_t)l * kD * kDM, Wtm1 + (size_t)l * kDM * kD, kD, kDM);
    transpose_bf16_kernel<<<cdiv((size_t)kDM * kD, 256), blk, 0, stream>>>(
        mlp_w2 + (size_t)l * kDM * kD, Wtm2 + (size_t)l * kD * kDM, kDM, kD);
  }
  transpose_bf16_kernel<<<cdiv((size_t)kD * 1024, 256), blk, 0, stream>>>(br_w1, WtB1, kD, 1024);
  transpose_bf16_kernel<<<cdiv((size_t)1024 * kDO, 256), blk, 0, stream>>>(br_w2, WtB2, 1024, kDO);

  pos_kernel<<<cdiv((size_t)256 * kD, 256), blk, 0, stream>>>(row_emb, col_emb, posb);
  cls_kernel<<<cdiv((size_t)kB * kD, 256), blk, 0, stream>>>(cls, x);
  patchify_kernel<<<cdiv((size_t)kMP * kKP, 256), blk, 0, stream>>>(pixel, A0);

  // patch embedding GEMM (adds bias + pos, scatters into x rows 1..256)
  gemm_wmma_kernel<4><<<dim3(kMP / 128, kD / 128), blk, 0, stream>>>(
      A0, WtP, patch_b, x, nullptr, nullptr, posb, kMP, kD, kKP);

  const unsigned gM = cdiv(kM, 128);
  for (int l = 0; l < kNL; ++l) {
    layernorm_kernel<false><<<kM / 8, blk, 0, stream>>>(x, ln1_s + l * kD, ln1_b + l * kD, nullptr, xn, kM);
    gemm_wmma_kernel<1><<<dim3(gM, (3 * kD) / 128), blk, 0, stream>>>(
        xn, Wtq + (size_t)l * 3 * kD * kD, qkv_b + (size_t)l * 3 * kD,
        nullptr, qkvb, nullptr, nullptr, kM, 3 * kD, kD);
    attn_kernel<<<kB * kNH, blk, 0, stream>>>(qkvb, atno);
    gemm_wmma_kernel<3><<<dim3(gM, kD / 128), blk, 0, stream>>>(
        atno, Wto + (size_t)l * kD * kD, out_b + (size_t)l * kD,
        x, nullptr, x, nullptr, kM, kD, kD);
    layernorm_kernel<false><<<kM / 8, blk, 0, stream>>>(x, ln2_s + l * kD, ln2_b + l * kD, nullptr, xn, kM);
    gemm_wmma_kernel<2><<<dim3(gM, kDM / 128), blk, 0, stream>>>(
        xn, Wtm1 + (size_t)l * kDM * kD, mlp_b1 + (size_t)l * kDM,
        nullptr, hb, nullptr, nullptr, kM, kDM, kD);
    gemm_wmma_kernel<3><<<dim3(gM, kD / 128), blk, 0, stream>>>(
        hb, Wtm2 + (size_t)l * kD * kDM, mlp_b2 + (size_t)l * kD,
        x, nullptr, x, nullptr, kM, kD, kDM);
  }

  // final LN + bridge MLP + bridge LN -> d_out (f32), then vis_mask (int32 ones)
  layernorm_kernel<false><<<kM / 8, blk, 0, stream>>>(x, lnf_s, lnf_b, nullptr, xn, kM);
  gemm_wmma_kernel<2><<<dim3(gM, 1024 / 128), blk, 0, stream>>>(
      xn, WtB1, br_b1, nullptr, h1, nullptr, nullptr, kM, 1024, kD);
  gemm_wmma_kernel<0><<<dim3(gM, kDO / 128), blk, 0, stream>>>(
      h1, WtB2, br_b2, h2, nullptr, nullptr, nullptr, kM, kDO, 1024);
  layernorm_kernel<true><<<kM / 8, blk, 0, stream>>>(h2, brln_s, brln_b, (float*)d_out, nullptr, kM);
  vismask_kernel<<<cdiv((size_t)kM, 256), blk, 0, stream>>>((int*)d_out + (size_t)kM * kDO, kM);
}
